// ann2_snn1_49185965474515
// MI455X (gfx1250) — matrix operand order repaired and checked by reading, not compiled, not run on hardware
//
#include <hip/hip_runtime.h>
#include <hip/hip_bf16.h>

typedef __attribute__((ext_vector_type(16))) _Float16 v16h;
typedef __attribute__((ext_vector_type(8)))  float    v8f;

#define B_    64
#define F_    500
#define FP_   512      // features padded to 16 chunks of 32
#define T_    1024
#define O_    10
#define OP_   16       // outputs padded to WMMA N
#define NSEG_ 16
#define L_    64       // T_ = NSEG_ * L_
#define NFC_  16       // feature chunks of 32
#define NBT_  4        // batch tiles of 16

// Fast sigmoid: v_mul (log2e fold) + v_exp_f32 + v_add + v_rcp_f32.
__device__ __forceinline__ float sigm(float x) {
  return __builtin_amdgcn_rcpf(1.0f + __expf(-x));
}

// Load the W^T chunk for feature-chunk fc as the WMMA B operand (32xK x 16xN, f16).
// Layout (matches A): n=lane&15, k=(j&7) + ((j>>3)<<4) + 8*(lane>>4).
// Padded features and padded outputs are exact zeros => padded A columns contribute 0.
__device__ __forceinline__ v16h load_bfrag(const float* __restrict__ W, int fc, int n, int hf) {
  v16h bf;
#pragma unroll
  for (int j = 0; j < 16; ++j) {
    int kb = (j & 7) + ((j >> 3) << 4) + (hf << 3);
    int fb = fc * 32 + kb;
    float w = (n < O_ && fb < F_) ? W[n * F_ + fb] : 0.0f;
    bf[j] = (_Float16)w;
  }
  return bf;
}

// K0: zero the psc accumulation buffer (atomics land here).
__global__ void k_zero(float4* __restrict__ p, int n4) {
  int i = blockIdx.x * blockDim.x + threadIdx.x;
  if (i < n4) p[i] = make_float4(0.f, 0.f, 0.f, 0.f);
}

// K1: zero-init segment IIR + fused WMMA projection + atomic psc accumulation,
// and emit the segment's final two states (y[L-1], y[L-2]).
// Padded streams (f >= F_) run with ca=cb=0 on aliased input data: y stays a finite
// sigmoid value in (0,1) and multiplies a zero B row, so no masking in the hot loop.
__global__ void __launch_bounds__(32)
k_seg_iir_proj(const float* __restrict__ in, const float* __restrict__ a1,
               const float* __restrict__ a2, const float* __restrict__ W,
               float* __restrict__ psc, float* __restrict__ pend) {
  const int fc  = blockIdx.x, bt = blockIdx.y, seg = blockIdx.z;
  const int lane = threadIdx.x;
  const int n    = lane & 15;
  const int hf   = lane >> 4;

  const v16h bfrag = load_bfrag(W, fc, n, hf);

  // Per-lane streams in the f16 A-matrix layout: this lane owns batch row
  // b = bt*16 + n, and 16 features k = (j&7) + ((j>>3)<<4) + hf*8.
  const int b = bt * 16 + n;
  float y1[16], y2[16], ca[16], cb[16];
  unsigned off[16];
#pragma unroll
  for (int j = 0; j < 16; ++j) {
    int k = (j & 7) + ((j >> 3) << 4) + (hf << 3);
    int f = fc * 32 + k;
    bool v = (f < F_);
    ca[j]  = v ? a1[f] : 0.0f;
    cb[j]  = v ? a2[f] : 0.0f;
    off[j] = v ? (unsigned)((b * F_ + f) * T_ + seg * L_) : 0u;
    y1[j] = 0.0f; y2[j] = 0.0f;
  }

  const int t0 = seg * L_;
  for (int tt = 0; tt < L_; tt += 4) {
    float xq[16][4];
#pragma unroll
    for (int j = 0; j < 16; ++j) {
      const float4 q = *(const float4*)(in + off[j] + tt);
      xq[j][0] = q.x; xq[j][1] = q.y; xq[j][2] = q.z; xq[j][3] = q.w;
    }
#pragma unroll
    for (int u = 0; u < 4; ++u) {
      v16h afrag;
#pragma unroll
      for (int j = 0; j < 16; ++j) {
        float x = sigm(xq[j][u]);
        float y = fmaf(ca[j], y1[j], fmaf(cb[j], y2[j], x));
        y2[j] = y1[j]; y1[j] = y;
        afrag[j] = (_Float16)y;
      }
      v8f acc = {};
      acc = __builtin_amdgcn_wmma_f32_16x16x32_f16(false, afrag, false, bfrag,
                                                   (short)0, acc, false, false);
      const int t = t0 + tt + u;
      float* base = psc + (size_t)t * (B_ * OP_);
      if (n < O_) {
#pragma unroll
        for (int r = 0; r < 8; ++r) {
          int bb = bt * 16 + hf * 8 + r;  // D layout: M = r + 8*hf, N = lane&15
          atomicAdd(base + bb * OP_ + n, acc[r]);
        }
      }
    }
  }

  // Zero-init end state for the inter-segment scan (mask padded streams here only).
#pragma unroll
  for (int j = 0; j < 16; ++j) {
    int k = (j & 7) + ((j >> 3) << 4) + (hf << 3);
    int f = fc * 32 + k;
    float msk = (f < F_) ? 1.0f : 0.0f;
    size_t ix = (((size_t)seg * B_ + b) * FP_ + f) * 2;
    pend[ix]     = y1[j] * msk;
    pend[ix + 1] = y2[j] * msk;
  }
}

// K2: sequential scan over the 16 segments per (b,f) stream.
// Segment transfer: (y[L-1],y[L-2]) = M^L * init + p_seg, with M^L built from the
// homogeneous responses alpha/beta iterated L_ steps (general per-feature coeffs).
__global__ void k_state_scan(const float* __restrict__ a1, const float* __restrict__ a2,
                             const float* __restrict__ pend, float* __restrict__ sinit) {
  int idx = blockIdx.x * blockDim.x + threadIdx.x;
  if (idx >= B_ * FP_) return;
  int b = idx / FP_, f = idx % FP_;
  float A1 = (f < F_) ? a1[f] : 0.0f;
  float A2 = (f < F_) ? a2[f] : 0.0f;
  float al1 = 1.f, al2 = 0.f, be1 = 0.f, be2 = 1.f;
  for (int t = 0; t < L_; ++t) {
    float na = fmaf(A1, al1, A2 * al2); al2 = al1; al1 = na;
    float nb = fmaf(A1, be1, A2 * be2); be2 = be1; be1 = nb;
  }
  float s1 = 0.f, s2 = 0.f;  // true (y[-1], y[-2]) entering segment s
  for (int s = 0; s < NSEG_; ++s) {
    size_t ix = (((size_t)s * B_ + b) * FP_ + f) * 2;
    sinit[ix] = s1; sinit[ix + 1] = s2;
    float p1 = pend[ix], p2 = pend[ix + 1];
    float e1 = fmaf(al1, s1, fmaf(be1, s2, p1));
    float e2 = fmaf(al2, s1, fmaf(be2, s2, p2));
    s1 = e1; s2 = e2;
  }
}

// K3: correction pass. The correction is the *homogeneous* recurrence seeded with
// each segment's true initial state -- no input re-read. Project with WMMA and
// atomically add into psc. Segment 0 has zero init => skipped (grid.z = NSEG_-1).
__global__ void __launch_bounds__(32)
k_seg_correct(const float* __restrict__ a1, const float* __restrict__ a2,
              const float* __restrict__ W, const float* __restrict__ sinit,
              float* __restrict__ psc) {
  const int fc  = blockIdx.x, bt = blockIdx.y, seg = blockIdx.z + 1;
  const int lane = threadIdx.x;
  const int n    = lane & 15;
  const int hf   = lane >> 4;

  const v16h bfrag = load_bfrag(W, fc, n, hf);

  const int b = bt * 16 + n;
  float h1[16], h2[16], ca[16], cb[16];
#pragma unroll
  for (int j = 0; j < 16; ++j) {
    int k = (j & 7) + ((j >> 3) << 4) + (hf << 3);
    int f = fc * 32 + k;
    bool v = (f < F_);
    ca[j] = v ? a1[f] : 0.0f;
    cb[j] = v ? a2[f] : 0.0f;
    size_t ix = (((size_t)seg * B_ + b) * FP_ + f) * 2;
    h1[j] = sinit[ix];      // padded entries are written as exact zeros by K2
    h2[j] = sinit[ix + 1];
  }

  const int t0 = seg * L_;
  for (int tau = 0; tau < L_; ++tau) {
    v16h afrag;
#pragma unroll
    for (int j = 0; j < 16; ++j) {
      float y = fmaf(ca[j], h1[j], cb[j] * h2[j]);
      h2[j] = h1[j]; h1[j] = y;
      afrag[j] = (_Float16)y;
    }
    v8f acc = {};
    acc = __builtin_amdgcn_wmma_f32_16x16x32_f16(false, afrag, false, bfrag,
                                                 (short)0, acc, false, false);
    float* base = psc + (size_t)(t0 + tau) * (B_ * OP_);
    if (n < O_) {
#pragma unroll
      for (int r = 0; r < 8; ++r) {
        int bb = bt * 16 + hf * 8 + r;
        atomicAdd(base + bb * OP_ + n, acc[r]);
      }
    }
  }
}

// K4: bias + LIF dynamics + spike threshold, 640 serial (b,o) streams.
__global__ void __launch_bounds__(640)
k_lif(const float* __restrict__ psc, const float* __restrict__ bias,
      float* __restrict__ out) {
  int idx = threadIdx.x;
  if (idx >= B_ * O_) return;
  int b = idx / O_, o = idx % O_;
  const float bo = bias[o];
  const float dm = 0.77880078307140487f;  // exp(-1/4)
  float v = 0.f, s = 0.f;
  float buf[4];
  float* op = out + ((size_t)b * O_ + o) * T_;
  for (int t = 0; t < T_; t += 4) {
#pragma unroll
    for (int u = 0; u < 4; ++u) {
      float it = psc[(size_t)(t + u) * (B_ * OP_) + b * OP_ + o] + bo;
      v = dm * v * (1.0f - s) + it;
      s = (v > 1.0f) ? 1.0f : 0.0f;
      buf[u] = s;
    }
    *(float4*)(op + t) = make_float4(buf[0], buf[1], buf[2], buf[3]);
  }
}

extern "C" void kernel_launch(void* const* d_in, const int* in_sizes, int n_in,
                              void* d_out, int out_size, void* d_ws, size_t ws_size,
                              hipStream_t stream) {
  (void)in_sizes; (void)n_in; (void)out_size; (void)ws_size;
  const float* in = (const float*)d_in[0];
  const float* a1 = (const float*)d_in[1];
  const float* a2 = (const float*)d_in[2];
  const float* W  = (const float*)d_in[3];
  const float* bs = (const float*)d_in[4];
  float* out = (float*)d_out;

  // Workspace layout (floats): psc [T,B,16] | pend [NSEG,B,FP,2] | sinit [NSEG,B,FP,2]
  float* ws   = (float*)d_ws;
  float* psc  = ws;
  float* pend = psc  + (size_t)T_ * B_ * OP_;
  float* sini = pend + (size_t)NSEG_ * B_ * FP_ * 2;

  const int n4 = (T_ * B_ * OP_) / 4;
  k_zero<<<dim3((n4 + 255) / 256), dim3(256), 0, stream>>>((float4*)psc, n4);
  k_seg_iir_proj<<<dim3(NFC_, NBT_, NSEG_), dim3(32), 0, stream>>>(in, a1, a2, W, psc, pend);
  k_state_scan<<<dim3((B_ * FP_ + 255) / 256), dim3(256), 0, stream>>>(a1, a2, pend, sini);
  k_seg_correct<<<dim3(NFC_, NBT_, NSEG_ - 1), dim3(32), 0, stream>>>(a1, a2, W, sini, psc);
  k_lif<<<dim3(1), dim3(640), 0, stream>>>(psc, bs, out);
}
